// MambaLayer_39548058862120
// MI455X (gfx1250) — matrix operand order repaired and checked by reading, not compiled, not run on hardware
//
#include <hip/hip_runtime.h>
#include <hip/hip_bf16.h>

// ---------------- types ----------------
typedef __attribute__((ext_vector_type(16))) __bf16 v16bf;
typedef __attribute__((ext_vector_type(8)))  __bf16 v8bf;
typedef __attribute__((ext_vector_type(8)))  float  v8f;

// Mamba dims (fixed by reference)
#define MB_B   2
#define MB_L   2048
#define MB_D   2048
#define MB_DIN 4096
#define MB_N   16
#define MB_R   128
#define MB_XD  (MB_R + 2 * MB_N)   // 160

__device__ __forceinline__ __bf16 f2bf(float f) {
    unsigned int u = __builtin_bit_cast(unsigned int, f);
    unsigned int r = u + 0x7FFFu + ((u >> 16) & 1u);   // round-to-nearest-even
    unsigned short h = (unsigned short)(r >> 16);
    return __builtin_bit_cast(__bf16, h);
}

// -------- fp32 -> bf16 streaming conversion --------
__global__ void cvt_f32_bf16_kernel(const float* __restrict__ in,
                                    __bf16* __restrict__ out, int n) {
    int i = blockIdx.x * blockDim.x + threadIdx.x;
    if (i < n) out[i] = f2bf(in[i]);
}

// -------- register-blocked, double-buffered NT WMMA GEMM --------
// C[M,N](f32) = A[M,K](bf16, row-major, stride lda) * W[N,K](bf16, row-major)^T
// Wave tile: 64 rows x 32 cols (4 m-tiles x 2 n-tiles, 8 WMMAs / k-step of 32).
// Block: 4 waves side by side in N -> block tile 64 x 128.
// K must be a multiple of 64 (true for all uses: 2048, 4096, 128).
// EPI: 0 = plain store, 1 = softplus(acc + bias[col])

struct Frags { v16bf a[4]; v16bf b[2]; };

__device__ __forceinline__ void load_frags(Frags& f,
                                           const __bf16* const (&arow)[4],
                                           const __bf16* const (&brow)[2],
                                           int k0, int half)
{
    // A 16x32 bf16 layout (ISA 7.12.2): half 0 -> K k0+0..7 & +16..23,
    //                                   half 1 -> K k0+8..15 & +24..31
    const int ka = k0 + half * 8;
    #pragma unroll
    for (int tm = 0; tm < 4; ++tm) {
        v8bf a0 = *(const v8bf*)(arow[tm] + ka);
        v8bf a1 = *(const v8bf*)(arow[tm] + ka + 16);
        f.a[tm] = __builtin_shufflevector(a0, a1, 0, 1, 2, 3, 4, 5, 6, 7,
                                          8, 9, 10, 11, 12, 13, 14, 15);
    }
    // B 32x16 bf16 layout: lane = column; half 0 -> K k0+0..7 & +16..23, half 1 -> K k0+8..15 & +24..31
    const int kb = k0 + half * 8;
    #pragma unroll
    for (int tn = 0; tn < 2; ++tn) {
        v8bf b0 = *(const v8bf*)(brow[tn] + kb);
        v8bf b1 = *(const v8bf*)(brow[tn] + kb + 16);
        f.b[tn] = __builtin_shufflevector(b0, b1, 0, 1, 2, 3, 4, 5, 6, 7,
                                          8, 9, 10, 11, 12, 13, 14, 15);
    }
}

__device__ __forceinline__ void do_wmmas(v8f (&acc)[4][2], const Frags& f)
{
    #pragma unroll
    for (int tm = 0; tm < 4; ++tm)
        #pragma unroll
        for (int tn = 0; tn < 2; ++tn)
            acc[tm][tn] = __builtin_amdgcn_wmma_f32_16x16x32_bf16(
                false, f.a[tm], false, f.b[tn],
                (short)0, acc[tm][tn], false, false);
}

template<int EPI>
__global__ __launch_bounds__(128)
void wmma_gemm_bf16(const __bf16* __restrict__ A, const __bf16* __restrict__ W,
                    float* __restrict__ C, const float* __restrict__ bias,
                    int M, int N, int K, int lda, int ldc)
{
    const int wave = threadIdx.x >> 5;
    const int lane = threadIdx.x & 31;
    const int n0w  = blockIdx.x * 128 + wave * 32;   // wave-uniform
    const int m0   = blockIdx.y * 64;
    if (n0w >= N) return;                            // wave-uniform: EXEC stays all-1s

    const int half = lane >> 4;                      // 0: lanes 0-15, 1: lanes 16-31
    const int lrow = lane & 15;

    const __bf16* arow[4];
    #pragma unroll
    for (int tm = 0; tm < 4; ++tm)
        arow[tm] = A + (size_t)(m0 + tm * 16 + lrow) * (size_t)lda;
    const __bf16* brow[2];
    #pragma unroll
    for (int tn = 0; tn < 2; ++tn)
        brow[tn] = W + (size_t)(n0w + tn * 16 + lrow) * (size_t)K;

    v8f acc[4][2] = {};
    Frags f0, f1;

    // software pipeline: fragments for k and k+32 are in flight while WMMAs run
    load_frags(f0, arow, brow, 0, half);
    for (int k0 = 0; k0 < K; k0 += 64) {
        load_frags(f1, arow, brow, k0 + 32, half);
        do_wmmas(acc, f0);
        if (k0 + 64 < K)
            load_frags(f0, arow, brow, k0 + 64, half);
        do_wmmas(acc, f1);
    }

    // C 16x16 f32 layout: VGPR j -> row tile_m0 + j + half*8, col tile_n0 + lrow
    #pragma unroll
    for (int tm = 0; tm < 4; ++tm) {
        #pragma unroll
        for (int tn = 0; tn < 2; ++tn) {
            const int col = n0w + tn * 16 + lrow;
            #pragma unroll
            for (int j = 0; j < 8; ++j) {
                const int row = m0 + tm * 16 + j + half * 8;
                float v = acc[tm][tn][j];
                if (EPI == 1) {
                    v += bias[col];
                    v = (v > 20.f) ? v : log1pf(expf(v));   // softplus
                }
                C[(size_t)row * (size_t)ldc + col] = v;
            }
        }
    }
}

// -------- depthwise causal conv1d (K=4) + SiLU; writes fp32 + bf16 --------
__global__ void conv_silu_kernel(const float* __restrict__ xz,
                                 const float* __restrict__ w4,
                                 float* __restrict__ xc,
                                 __bf16* __restrict__ xcb)
{
    size_t i = (size_t)blockIdx.x * blockDim.x + threadIdx.x;
    const size_t total = (size_t)MB_B * MB_L * MB_DIN;
    if (i >= total) return;
    const int d  = (int)(i % MB_DIN);
    const size_t bl = i / MB_DIN;
    const int l  = (int)(bl % MB_L);
    const int b  = (int)(bl / MB_L);

    const float* xb = xz + (size_t)b * MB_L * (2 * MB_DIN);
    float acc = 0.f;
    #pragma unroll
    for (int k = 0; k < 4; ++k) {
        const int ls = l + k - 3;
        if (ls >= 0) acc += w4[d * 4 + k] * xb[(size_t)ls * (2 * MB_DIN) + d];
    }
    const float s = acc / (1.f + expf(-acc));      // SiLU
    xc[i]  = s;
    xcb[i] = f2bf(s);
}

// -------- selective scan, fused epilogue: y = (scan + xc*D) * silu(z) --------
__global__ __launch_bounds__(256)
void scan_kernel(const float* __restrict__ delta,
                 const float* __restrict__ xdbl,   // (B*L, 160): [dt(128)|B(16)|C(16)]
                 const float* __restrict__ xc,
                 const float* __restrict__ xz,     // z = cols [DIN, 2*DIN)
                 const float* __restrict__ A_log,
                 const float* __restrict__ Dp,
                 __bf16* __restrict__ y)
{
    const int CH = 32;                             // timesteps staged per LDS chunk
    __shared__ float sB[CH * MB_N];
    __shared__ float sC[CH * MB_N];

    const int b = blockIdx.y;
    const int d = blockIdx.x * 256 + threadIdx.x;

    float Arow[MB_N], h[MB_N];
    #pragma unroll
    for (int n = 0; n < MB_N; ++n) {
        Arow[n] = -expf(A_log[(size_t)d * MB_N + n]);
        h[n] = 0.f;
    }
    const float dpar = Dp[d];

    for (int l0 = 0; l0 < MB_L; l0 += CH) {
        __syncthreads();
        for (int idx = threadIdx.x; idx < CH * MB_N; idx += 256) {
            const int lo = idx >> 4, n = idx & 15;
            const size_t base = ((size_t)b * MB_L + (l0 + lo)) * MB_XD;
            sB[idx] = xdbl[base + MB_R + n];
            sC[idx] = xdbl[base + MB_R + MB_N + n];
        }
        __syncthreads();

        for (int lo = 0; lo < CH; ++lo) {
            const int l = l0 + lo;
            const size_t off = ((size_t)b * MB_L + l) * MB_DIN + d;
            const float dt  = delta[off];
            const float xv  = xc[off];
            const float dtx = dt * xv;
            float yv = 0.f;
            #pragma unroll
            for (int n = 0; n < MB_N; ++n) {
                const float dA = expf(dt * Arow[n]);
                h[n] = h[n] * dA + dtx * sB[lo * MB_N + n];
                yv += h[n] * sC[lo * MB_N + n];
            }
            yv += xv * dpar;
            const float zv = xz[((size_t)b * MB_L + l) * (2 * MB_DIN) + MB_DIN + d];
            yv *= zv / (1.f + expf(-zv));          // * silu(z)
            y[off] = f2bf(yv);
        }
    }
}

// ---------------- host-side orchestration ----------------
extern "C" void kernel_launch(void* const* d_in, const int* in_sizes, int n_in,
                              void* d_out, int out_size, void* d_ws, size_t ws_size,
                              hipStream_t stream)
{
    const float* hidden   = (const float*)d_in[0];  // (B,L,D)
    const float* in_w     = (const float*)d_in[1];  // (2*DIN, D)
    const float* conv_w   = (const float*)d_in[2];  // (DIN, 4)
    const float* xp_w     = (const float*)d_in[3];  // (R+2N, DIN)
    const float* dt_w     = (const float*)d_in[4];  // (DIN, R)
    const float* dt_bias  = (const float*)d_in[5];  // (DIN,)
    const float* A_log    = (const float*)d_in[6];  // (DIN, N)
    const float* D_param  = (const float*)d_in[7];  // (DIN,)
    const float* out_w    = (const float*)d_in[8];  // (D, DIN)
    float* out            = (float*)d_out;          // (B,L,D) fp32

    const int Mtok = MB_B * MB_L;                   // 4096 token rows

    // workspace carve-up (bytes)
    char* ws = (char*)d_ws;
    size_t off = 0;
    auto take = [&](size_t bytes) { char* p = ws + off; off += (bytes + 255) & ~(size_t)255; return p; };

    float*  xz      = (float*)take((size_t)Mtok * 2 * MB_DIN * 4);         // 134 MB
    float*  xc      = (float*)take((size_t)Mtok * MB_DIN * 4);             //  67 MB
    float*  xdbl    = (float*)take((size_t)Mtok * MB_XD * 4);              // 2.6 MB
    float*  delta   = (float*)take((size_t)Mtok * MB_DIN * 4);             //  67 MB
    __bf16* hid_bf  = (__bf16*)take((size_t)Mtok * MB_D * 2);              //  17 MB
    __bf16* xc_bf   = (__bf16*)take((size_t)Mtok * MB_DIN * 2);            //  34 MB
    __bf16* xdbl_bf = (__bf16*)take((size_t)Mtok * MB_XD * 2);             // 1.3 MB
    __bf16* y_bf    = (__bf16*)take((size_t)Mtok * MB_DIN * 2);            //  34 MB
    __bf16* w_in    = (__bf16*)take((size_t)2 * MB_DIN * MB_D * 2);        //  34 MB
    __bf16* w_xp    = (__bf16*)take((size_t)MB_XD * MB_DIN * 2);
    __bf16* w_dt    = (__bf16*)take((size_t)MB_DIN * MB_R * 2);
    __bf16* w_out   = (__bf16*)take((size_t)MB_D * MB_DIN * 2);            //  17 MB

    // 1) one-time bf16 conversions: weights + hidden activations
    {
        int n1 = 2 * MB_DIN * MB_D;
        cvt_f32_bf16_kernel<<<(n1 + 255) / 256, 256, 0, stream>>>(in_w, w_in, n1);
        int n2 = MB_XD * MB_DIN;
        cvt_f32_bf16_kernel<<<(n2 + 255) / 256, 256, 0, stream>>>(xp_w, w_xp, n2);
        int n3 = MB_DIN * MB_R;
        cvt_f32_bf16_kernel<<<(n3 + 255) / 256, 256, 0, stream>>>(dt_w, w_dt, n3);
        int n4 = MB_D * MB_DIN;
        cvt_f32_bf16_kernel<<<(n4 + 255) / 256, 256, 0, stream>>>(out_w, w_out, n4);
        int n5 = Mtok * MB_D;
        cvt_f32_bf16_kernel<<<(n5 + 255) / 256, 256, 0, stream>>>(hidden, hid_bf, n5);
    }

    // 2) in_proj: xz[Mtok, 8192] = hidden * in_w^T
    {
        dim3 grid((2 * MB_DIN + 127) / 128, Mtok / 64);
        wmma_gemm_bf16<0><<<grid, 128, 0, stream>>>(hid_bf, w_in, xz, nullptr,
                                                    Mtok, 2 * MB_DIN, MB_D, MB_D, 2 * MB_DIN);
    }

    // 3) conv1d + silu -> xc (fp32) and xc_bf (bf16)
    {
        size_t total = (size_t)Mtok * MB_DIN;
        conv_silu_kernel<<<(unsigned)((total + 255) / 256), 256, 0, stream>>>(xz, conv_w, xc, xc_bf);
    }

    // 4) x_proj: xdbl[Mtok, 160] = xc * xp_w^T
    {
        dim3 grid((MB_XD + 127) / 128, Mtok / 64);
        wmma_gemm_bf16<0><<<grid, 128, 0, stream>>>(xc_bf, w_xp, xdbl, nullptr,
                                                    Mtok, MB_XD, MB_DIN, MB_DIN, MB_XD);
    }

    // 5) bf16 copy of x_dbl (dt columns feed dt_proj GEMM with lda=160)
    {
        int n = Mtok * MB_XD;
        cvt_f32_bf16_kernel<<<(n + 255) / 256, 256, 0, stream>>>(xdbl, xdbl_bf, n);
    }

    // 6) dt_proj + bias + softplus: delta = softplus(xdbl[:, :128] * dt_w^T + bias)
    {
        dim3 grid((MB_DIN + 127) / 128, Mtok / 64);
        wmma_gemm_bf16<1><<<grid, 128, 0, stream>>>(xdbl_bf, w_dt, delta, dt_bias,
                                                    Mtok, MB_DIN, MB_R, MB_XD, MB_DIN);
    }

    // 7) selective scan + skip + gate -> y_bf (bf16)
    {
        dim3 grid(MB_DIN / 256, MB_B);
        scan_kernel<<<grid, 256, 0, stream>>>(delta, xdbl, xc, xz, A_log, D_param, y_bf);
    }

    // 8) out_proj: out[Mtok, 2048] = y * out_w^T
    {
        dim3 grid((MB_D + 127) / 128, Mtok / 64);
        wmma_gemm_bf16<0><<<grid, 128, 0, stream>>>(y_bf, w_out, out, nullptr,
                                                    Mtok, MB_D, MB_DIN, MB_DIN, MB_D);
    }
}
